// CustomLSTM_26147760898762
// MI455X (gfx1250) — matrix operand order repaired and checked by reading, not compiled, not run on hardware
//
#include <hip/hip_runtime.h>
#include <hip/hip_bf16.h>
#include <math.h>

// Bidirectional LSTM for MI455X (gfx1250, wave32, WMMA).
// Per-timestep fused GEMM: gates = [x_t ; h_{t-1}] (K=1536) x Wcat^T + bias.
// - f16 WMMA (16x16x32) with fp32 accumulate.
// - One workgroup per (dir, jblock) covers ALL 64 batch rows: the 4 batch-tile
//   waves per gate read identical weight addresses -> WGP$ hits, ~4x less L2
//   weight traffic than spreading batch tiles across WGPs.
// - 16 waves/WGP (4 per SIMD32) + double-buffered fragment prefetch to hide
//   L2 latency (round-1 asm showed s_wait_loadcnt 0 right before each wmma).
// - h state ping-pongs between buffers across the 512 stream-ordered launches.

#define Bb 64
#define Tt 512
#define Ii 512
#define Hh 1024
#define G4 4096   // 4*H
#define K4 1536   // I + H
#define KC 48     // K4 / 32

typedef _Float16 v16h __attribute__((ext_vector_type(16)));
typedef _Float16 v8h  __attribute__((ext_vector_type(8)));
typedef float    v8f  __attribute__((ext_vector_type(8)));

// ---------------- prep kernels ----------------

// Wcat[dir][row(4H)][k(K4)] f16, row-major: k<I from w_ih, else w_hh.
__global__ __launch_bounds__(256)
void prep_weights(const float* __restrict__ wihf, const float* __restrict__ whhf,
                  const float* __restrict__ wihb, const float* __restrict__ whhb,
                  _Float16* __restrict__ Wcat) {
  int r   = blockIdx.x;          // 0..8191
  int dir = r >> 12;
  int row = r & 4095;
  const float* wih = dir ? wihb : wihf;
  const float* whh = dir ? whhb : whhf;
  _Float16* dst = Wcat + (size_t)r * K4;
  for (int k = threadIdx.x; k < K4; k += 256) {
    float v = (k < Ii) ? wih[(size_t)row * Ii + k]
                       : whh[(size_t)row * Hh + (k - Ii)];
    dst[k] = (_Float16)v;
  }
}

__global__ __launch_bounds__(256)
void prep_x(const float* __restrict__ x, _Float16* __restrict__ x_h, int n) {
  for (int i = blockIdx.x * blockDim.x + threadIdx.x; i < n;
       i += gridDim.x * blockDim.x)
    x_h[i] = (_Float16)x[i];
}

// bias[dir][4H] = b_ih + b_hh ; zero h ping buffer 0 (both dirs) and c state.
__global__ __launch_bounds__(256)
void prep_misc(const float* __restrict__ bihf, const float* __restrict__ bhhf,
               const float* __restrict__ bihb, const float* __restrict__ bhhb,
               float* __restrict__ bias, _Float16* __restrict__ h_h,
               float* __restrict__ c_st) {
  int i = blockIdx.x * blockDim.x + threadIdx.x;   // 0..131071
  if (i < 2 * G4) {
    int dir = i >> 12; int g = i & 4095;
    bias[i] = dir ? (bihb[g] + bhhb[g]) : (bihf[g] + bhhf[g]);
  }
  if (i < 2 * Bb * Hh) {       // buffer 0, both directions
    h_h[i]  = (_Float16)0.f;
    c_st[i] = 0.f;
  }
}

// ---------------- per-timestep kernel ----------------
// grid: 32 blocks = dir(2) x jblock(16, 64 h-cols each)
// block: 512 threads = 16 waves = gate(4) x mtile(4, 16 batch rows each).
// Waves with the same gate but different mtile fetch identical B fragments
// (WGP$ reuse); each wave owns a 16x64 fp32 accumulator tile.
__global__ __launch_bounds__(512)
void lstm_step(const _Float16* __restrict__ x_h, const _Float16* __restrict__ Wcat,
               const float* __restrict__ bias, _Float16* __restrict__ h_h,
               float* __restrict__ c_st, float* __restrict__ out, int t) {
  __shared__ float gsm[4][4][16][64];   // [mtile][gate][m][n] = 64 KB

  const int tid  = threadIdx.x;
  const int wid  = tid >> 5;
  const int gate = wid & 3;
  const int mt   = wid >> 2;          // batch tile 0..3
  const int lane = tid & 31;
  const int ln16 = lane & 15;
  const int hi   = lane >> 4;         // 0/1: lane half
  const int abase = hi * 8;           // A-frag K sub-offset per ISA layout

  const int bx     = blockIdx.x;
  const int dir    = bx >> 4;
  const int jblock = bx & 15;
  const int b0     = mt * 16;
  const int jbase  = jblock * 64;
  const int t_x    = dir ? (Tt - 1 - t) : t;   // bw consumes time-reversed x

  const _Float16* Wd  = Wcat + (size_t)dir * G4 * K4;
  const _Float16* hrd = h_h + ((size_t)((t & 1) * 2 + dir)) * Bb * Hh;        // read
  _Float16*       hwr = h_h + ((size_t)(((t & 1) ^ 1) * 2 + dir)) * Bb * Hh;  // write

  // accumulators, pre-loaded with combined bias (broadcast over batch rows)
  v8f acc[4];
#pragma unroll
  for (int nt = 0; nt < 4; ++nt) {
    float bv = bias[dir * G4 + gate * Hh + jbase + nt * 16 + ln16];
#pragma unroll
    for (int r = 0; r < 8; ++r) acc[nt][r] = bv;
  }

  const int    m    = b0 + ln16;                       // batch row this lane feeds
  const size_t xrow = ((size_t)m * Tt + t_x) * Ii;
  const size_t hrow = (size_t)m * Hh;
  const size_t wrow = (size_t)(gate * Hh + jbase + ln16) * K4;

  // fragment loaders (kc indexes 32-wide K chunks over the unified K=1536)
  auto loadA = [&](int kc) -> v16h {
    const _Float16* p = (kc < 16)
        ? (x_h + xrow + kc * 32 + abase)
        : (hrd + hrow + (kc - 16) * 32 + abase);
    union { v16h v; v8h h[2]; } u;
    u.h[0] = *(const v8h*)p;
    u.h[1] = *(const v8h*)(p + 16);
    return u.v;
  };
  auto loadB = [&](int kc, int nt) -> v16h {
    const _Float16* q = Wd + wrow + (size_t)nt * 16 * K4 + kc * 32 + hi * 8;
    union { v16h v; v8h h[2]; } u;
    u.h[0] = *(const v8h*)q;
    u.h[1] = *(const v8h*)(q + 16);
    return u.v;
  };

  // software-pipelined K loop: prefetch kc+1 while issuing WMMAs for kc
  v16h aF[2];
  v16h bF[2][4];
  aF[0] = loadA(0);
#pragma unroll
  for (int nt = 0; nt < 4; ++nt) bF[0][nt] = loadB(0, nt);

#pragma unroll 2
  for (int kc = 0; kc < KC; ++kc) {
    const int cur = kc & 1, nxt = cur ^ 1;
    if (kc + 1 < KC) {
      aF[nxt] = loadA(kc + 1);
#pragma unroll
      for (int nt = 0; nt < 4; ++nt) bF[nxt][nt] = loadB(kc + 1, nt);
    }
#pragma unroll
    for (int nt = 0; nt < 4; ++nt)
      acc[nt] = __builtin_amdgcn_wmma_f32_16x16x32_f16(
          false, aF[cur], false, bF[cur][nt], (short)0, acc[nt], false, false);
  }

  // dump this wave's gate tile (16x64 f32) to LDS; C layout: VGPR r -> M = r + 8*hi
#pragma unroll
  for (int nt = 0; nt < 4; ++nt)
#pragma unroll
    for (int r = 0; r < 8; ++r)
      gsm[mt][gate][r + abase][nt * 16 + ln16] = acc[nt][r];
  __syncthreads();

  // elementwise LSTM cell update: 64x64 = 4096 elems over 512 threads
  float* cdir = c_st + (size_t)dir * Bb * Hh;
#pragma unroll
  for (int e = 0; e < 8; ++e) {
    const int ii = e * 512 + tid;
    const int mtt = ii >> 10;
    const int rr  = ii & 1023;
    const int mm  = rr >> 6;
    const int nn  = rr & 63;
    const int b   = mtt * 16 + mm;
    const int j   = jbase + nn;
    const float gi = gsm[mtt][0][mm][nn];
    const float gf = gsm[mtt][1][mm][nn];
    const float gg = gsm[mtt][2][mm][nn];
    const float go = gsm[mtt][3][mm][nn];
    const size_t cidx = (size_t)b * Hh + j;
    const float cprev = cdir[cidx];
    const float si = 1.f / (1.f + __expf(-gi));
    const float sf = 1.f / (1.f + __expf(-gf));
    const float so = 1.f / (1.f + __expf(-go));
    const float cn = sf * cprev + si * tanhf(gg);
    const float hn = so * tanhf(cn);
    cdir[cidx] = cn;
    hwr[cidx]  = (_Float16)hn;
    out[(((size_t)b * Tt) + t) * (2 * Hh) + (size_t)dir * Hh + j] = hn;
    if (t == Tt - 1) {   // tail: h_fw, c_fw, h_bw, c_bw
      const size_t tail = (size_t)Bb * Tt * 2 * Hh;
      out[tail + (size_t)(2 * dir) * Bb * Hh + cidx]     = hn;
      out[tail + (size_t)(2 * dir + 1) * Bb * Hh + cidx] = cn;
    }
  }
}

// ---------------- launcher ----------------

extern "C" void kernel_launch(void* const* d_in, const int* in_sizes, int n_in,
                              void* d_out, int out_size, void* d_ws, size_t ws_size,
                              hipStream_t stream) {
  const float* x    = (const float*)d_in[0];
  const float* wihf = (const float*)d_in[1];
  const float* whhf = (const float*)d_in[2];
  const float* bihf = (const float*)d_in[3];
  const float* bhhf = (const float*)d_in[4];
  const float* wihb = (const float*)d_in[5];
  const float* whhb = (const float*)d_in[6];
  const float* bihb = (const float*)d_in[7];
  const float* bhhb = (const float*)d_in[8];
  float* out = (float*)d_out;

  char* ws = (char*)d_ws;
  size_t off = 0;
  _Float16* Wcat = (_Float16*)(ws + off); off += (size_t)2 * G4 * K4 * 2;      // 24 MB
  _Float16* x_h  = (_Float16*)(ws + off); off += (size_t)Bb * Tt * Ii * 2;     // 32 MB
  _Float16* h_h  = (_Float16*)(ws + off); off += (size_t)2 * 2 * Bb * Hh * 2;  // ping-pong x dirs
  float*    c_st = (float*)   (ws + off); off += (size_t)2 * Bb * Hh * 4;
  float*    bias = (float*)   (ws + off); off += (size_t)2 * G4 * 4;
  (void)off; (void)ws_size; (void)in_sizes; (void)n_in; (void)out_size;

  prep_weights<<<2 * G4, 256, 0, stream>>>(wihf, whhf, wihb, whhb, Wcat);
  prep_x<<<8192, 256, 0, stream>>>(x, x_h, Bb * Tt * Ii);
  prep_misc<<<512, 256, 0, stream>>>(bihf, bhhf, bihb, bhhb, bias, h_h, c_st);

  for (int t = 0; t < Tt; ++t)
    lstm_step<<<32, 512, 0, stream>>>(x_h, Wcat, bias, h_h, c_st, out, t);
}
